// MaskedCausalAttention_46969762349436
// MI455X (gfx1250) — matrix operand order repaired and checked by reading, not compiled, not run on hardware
//
#include <hip/hip_runtime.h>

typedef __bf16 bf16;
typedef __attribute__((ext_vector_type(16))) __bf16 bf16x16;
typedef __attribute__((ext_vector_type(8)))  __bf16 bf16x8;
typedef __attribute__((ext_vector_type(8)))  float   v8f;

#define B_  2
#define T_  2048
#define C_  1024
#define NH_ 16
#define D_  64
#define M_  (B_ * T_)   // 4096 rows for all projections

// ---------------------------------------------------------------------------
// Fragment loaders following CDNA5 ISA §7.12.2 VGPR layouts (wave32).
// A (16x32 bf16, M x K):  lane<16: row M=lane, K={0..7,16..23}
//                         lane>=16: row M=lane-16, K={8..15,24..31}
// B (32x16 bf16, K x N):  lanes 0-15: col N=lane,  K={0..7,16..23}
//                         lanes 16-31: col N=lane-16, K={8..15,24..31}
// C/D (16x16 f32): VGPR i: lanes 0-15 -> (M=i, N=lane); 16-31 -> (M=8+i).
// ---------------------------------------------------------------------------
__device__ __forceinline__ bf16x16 afrag(const bf16* base, int stride,
                                         int m_base, int k_base, int lane) {
  int m  = m_base + (lane & 15);
  int kh = (lane >> 4) << 3;                       // 0 or 8
  const bf16* p = base + (size_t)m * stride + k_base;
  bf16x8 lo = *reinterpret_cast<const bf16x8*>(p + kh);
  bf16x8 hi = *reinterpret_cast<const bf16x8*>(p + 16 + kh);
  bf16x16 r;
#pragma unroll
  for (int j = 0; j < 8; ++j) { r[j] = lo[j]; r[j + 8] = hi[j]; }
  return r;
}

__device__ __forceinline__ bf16x16 bfrag(const bf16* base, int stride,
                                         int n_base, int k_base, int lane) {
  int n  = n_base + (lane & 15);
  int ks = k_base + ((lane >> 4) << 3);            // 0 or 8
  const bf16* p = base + (size_t)n * stride + ks;
  bf16x8 lo = *reinterpret_cast<const bf16x8*>(p);
  bf16x8 hi = *reinterpret_cast<const bf16x8*>(p + 16);
  bf16x16 r;
#pragma unroll
  for (int j = 0; j < 8; ++j) { r[j] = lo[j]; r[j + 8] = hi[j]; }
  return r;
}

__device__ __forceinline__ v8f wmma_bf16(bf16x16 a, bf16x16 b, v8f c) {
  return __builtin_amdgcn_wmma_f32_16x16x32_bf16(false, a, false, b,
                                                 (short)0, c, false, false);
}

// Generic pointer -> 32-bit LDS offset (addrspace(3) pointers are 32-bit).
typedef __attribute__((address_space(3))) const void lds_cv;
__device__ __forceinline__ unsigned lds_off(const void* p) {
  return (unsigned)(unsigned long long)(lds_cv*)p;
}

// CDNA5 async copy global -> LDS, 16 bytes per lane, tracked by ASYNCcnt.
__device__ __forceinline__ void async_load16(const bf16* g, const bf16* l) {
  unsigned loff = lds_off(l);
  unsigned long long ga = (unsigned long long)(uintptr_t)g;
  asm volatile("global_load_async_to_lds_b128 %0, %1, off"
               :: "v"(loff), "v"(ga) : "memory");
}

// ---------------------------------------------------------------------------
// fp32 -> bf16 convert
// ---------------------------------------------------------------------------
__global__ void cvt_f32_bf16(const float* __restrict__ in,
                             bf16* __restrict__ out, int n) {
  int i = blockIdx.x * blockDim.x + threadIdx.x;
  if (i < n) out[i] = (bf16)in[i];
}

// ---------------------------------------------------------------------------
// GEMM (NT): out[m,n] = sum_k A[m,k]*B[n,k] + bias[n]   (M_=4096, N=K=1024)
// Block = 128 threads (4 waves) on a 128x64 tile; wave w owns rows 32w..32w+31.
// A/B chunks (K=32) staged in LDS via async-to-LDS, double buffered:
// the async engine fetches chunk k+1 while WMMAs consume chunk k from LDS.
// mode 0: fp32 row-major [M_,C_]            (final projection)
// mode 1: bf16 scatter to [B,H,T,D], scaled (Q with 1/sqrt(D), K)
// mode 2: bf16 scatter to [B,H,D,T]         (transposed V)
// ---------------------------------------------------------------------------
__global__ __launch_bounds__(128) void gemm_bf16nt(
    const bf16* __restrict__ A, const bf16* __restrict__ Bm,
    const float* __restrict__ bias, float scale, int mode,
    float* __restrict__ outf, bf16* __restrict__ outb) {
  __shared__ __align__(16) bf16 sA[2][128][32];   // 16 KB
  __shared__ __align__(16) bf16 sB[2][64][32];    //  8 KB

  const int tid   = threadIdx.x;
  const int lane  = tid & 31;
  const int wid   = tid >> 5;
  const int m_blk = blockIdx.x * 128;
  const int n_blk = blockIdx.y * 64;

  v8f acc[2][4] = {};

  auto issue = [&](int buf, int k) {
    // A tile 128x32: 512 16B-chunks, 4 per thread
#pragma unroll
    for (int it = 0; it < 4; ++it) {
      int idx = it * 128 + tid;
      int r = idx >> 2, c8 = (idx & 3) << 3;
      async_load16(A + (size_t)(m_blk + r) * C_ + k + c8, &sA[buf][r][c8]);
    }
    // B tile 64x32: 256 16B-chunks, 2 per thread
#pragma unroll
    for (int it = 0; it < 2; ++it) {
      int idx = it * 128 + tid;
      int r = idx >> 2, c8 = (idx & 3) << 3;
      async_load16(Bm + (size_t)(n_blk + r) * C_ + k + c8, &sB[buf][r][c8]);
    }
  };

  issue(0, 0);
  for (int kc = 0; kc < 32; ++kc) {
    const int cur = kc & 1;
    __syncthreads();                      // prior reads of buf cur^1 complete
    if (kc + 1 < 32) {
      issue(cur ^ 1, (kc + 1) * 32);      // prefetch next chunk
      asm volatile("s_wait_asynccnt 6" ::: "memory");   // cur chunk landed
    } else {
      asm volatile("s_wait_asynccnt 0" ::: "memory");
    }
    __syncthreads();                      // cur chunk visible to all waves
#pragma unroll
    for (int sm = 0; sm < 2; ++sm) {
      bf16x16 a = afrag(&sA[cur][0][0], 32, wid * 32 + sm * 16, 0, lane);
#pragma unroll
      for (int f = 0; f < 4; ++f) {
        bf16x16 b = bfrag(&sB[cur][0][0], 32, f * 16, 0, lane);
        acc[sm][f] = wmma_bf16(a, b, acc[sm][f]);
      }
    }
  }

  const int mh = lane >> 4;
  const int nl = lane & 15;
#pragma unroll
  for (int sm = 0; sm < 2; ++sm) {
#pragma unroll
    for (int f = 0; f < 4; ++f) {
      int n = n_blk + f * 16 + nl;
      float bval = bias[n];
#pragma unroll
      for (int i = 0; i < 8; ++i) {
        int m = m_blk + wid * 32 + sm * 16 + mh * 8 + i;
        float v = (acc[sm][f][i] + bval) * scale;
        if (mode == 0) {
          outf[(size_t)m * C_ + n] = v;
        } else {
          int b = m >> 11, t = m & (T_ - 1);   // T_ = 2048
          int h = n >> 6,  d = n & (D_ - 1);   // D_ = 64
          size_t idx = (mode == 1)
              ? (((size_t)(b * NH_ + h) * T_ + t) * D_ + d)
              : (((size_t)(b * NH_ + h) * D_ + d) * T_ + t);
          outb[idx] = (bf16)v;
        }
      }
    }
  }
}

// ---------------------------------------------------------------------------
// Causal flash attention. One wave per 16-query tile per head.
// Q pre-scaled by 1/sqrt(D). Streams 32-wide key chunks with online softmax.
// Software pipelined: V frags for chunk c and K frags for chunk c+1 are
// issued before the softmax VALU work so their latency hides behind it.
// Output: bf16 [B, T, C] (heads concatenated) for the final projection.
// ---------------------------------------------------------------------------
__global__ __launch_bounds__(128) void attn_flash(
    const bf16* __restrict__ Q, const bf16* __restrict__ K,
    const bf16* __restrict__ Vt, bf16* __restrict__ O) {
  __shared__ __align__(16) bf16 pbuf[4][16][32];

  const int lane  = threadIdx.x & 31;
  const int wid   = threadIdx.x >> 5;
  const int gw    = blockIdx.x * 4 + wid;
  const int ttile = gw & 127;                 // T_/16 = 128 tiles per head
  const int bh    = gw >> 7;                  // 0..B_*NH_-1
  const int t_base = ttile << 4;
  const size_t hoff = (size_t)bh * T_ * D_;   // T_*D_ == D_*T_ per head
  const bf16* Qh = Q + hoff;
  const bf16* Kh = K + hoff;
  const bf16* Vh = Vt + hoff;

  const bf16x16 aq0 = afrag(Qh, D_, t_base, 0, lane);
  const bf16x16 aq1 = afrag(Qh, D_, t_base, 32, lane);

  v8f o[4] = {};
  float mrun[8], lrun[8];
#pragma unroll
  for (int i = 0; i < 8; ++i) { mrun[i] = -3.0e38f; lrun[i] = 0.f; }

  const int mh = lane >> 4;
  const int nl = lane & 15;
  const int nchunks = (t_base + 47) >> 5;     // cover s <= t_base+15

  // K fragments for chunk 0 (register double buffer kb <- kn across loop)
  bf16x16 kb0 = bfrag(Kh, D_, 0, 0, lane);
  bf16x16 kb1 = bfrag(Kh, D_, 0, 32, lane);
  bf16x16 kb2 = bfrag(Kh, D_, 16, 0, lane);
  bf16x16 kb3 = bfrag(Kh, D_, 16, 32, lane);

  for (int c = 0; c < nchunks; ++c) {
    const int s_base = c << 5;

    // issue V(c) and K(c+1) loads up front; they complete under softmax
    bf16x16 vb0 = bfrag(Vh, T_, 0,  s_base, lane);
    bf16x16 vb1 = bfrag(Vh, T_, 16, s_base, lane);
    bf16x16 vb2 = bfrag(Vh, T_, 32, s_base, lane);
    bf16x16 vb3 = bfrag(Vh, T_, 48, s_base, lane);
    bf16x16 kn0, kn1, kn2, kn3;
    if (c + 1 < nchunks) {
      int sn = s_base + 32;
      kn0 = bfrag(Kh, D_, sn, 0, lane);
      kn1 = bfrag(Kh, D_, sn, 32, lane);
      kn2 = bfrag(Kh, D_, sn + 16, 0, lane);
      kn3 = bfrag(Kh, D_, sn + 16, 32, lane);
    }

    v8f s0 = {}, s1 = {};
    s0 = wmma_bf16(aq0, kb0, s0);
    s0 = wmma_bf16(aq1, kb1, s0);
    s1 = wmma_bf16(aq0, kb2, s1);
    s1 = wmma_bf16(aq1, kb3, s1);

    // causal mask: element (row t_base+mh*8+i, col s_base[+16]+nl)
#pragma unroll
    for (int i = 0; i < 8; ++i) {
      int t = t_base + mh * 8 + i;
      if (s_base + nl      > t) s0[i] = -3.0e38f;
      if (s_base + 16 + nl > t) s1[i] = -3.0e38f;
    }
    // online softmax (row reductions across the 16-lane fragment halves)
#pragma unroll
    for (int i = 0; i < 8; ++i) {
      float mx = fmaxf(s0[i], s1[i]);
#pragma unroll
      for (int m = 1; m <= 8; m <<= 1) mx = fmaxf(mx, __shfl_xor(mx, m, 32));
      float mnew  = fmaxf(mrun[i], mx);
      float alpha = __expf(mrun[i] - mnew);
      float p0 = __expf(s0[i] - mnew);
      float p1 = __expf(s1[i] - mnew);
      s0[i] = p0; s1[i] = p1;
      float rs = p0 + p1;
#pragma unroll
      for (int m = 1; m <= 8; m <<= 1) rs += __shfl_xor(rs, m, 32);
      lrun[i] = lrun[i] * alpha + rs;
      mrun[i] = mnew;
#pragma unroll
      for (int f = 0; f < 4; ++f) o[f][i] *= alpha;
    }
    // repack P (C layout) -> A layout through LDS (16x32 bf16 per wave)
#pragma unroll
    for (int i = 0; i < 8; ++i) {
      int r = mh * 8 + i;
      pbuf[wid][r][nl]      = (bf16)s0[i];
      pbuf[wid][r][16 + nl] = (bf16)s1[i];
    }
    asm volatile("s_wait_dscnt 0" ::: "memory");
    bf16x16 pa = afrag(&pbuf[wid][0][0], 32, 0, 0, lane);
    o[0] = wmma_bf16(pa, vb0, o[0]);
    o[1] = wmma_bf16(pa, vb1, o[1]);
    o[2] = wmma_bf16(pa, vb2, o[2]);
    o[3] = wmma_bf16(pa, vb3, o[3]);

    if (c + 1 < nchunks) { kb0 = kn0; kb1 = kn1; kb2 = kn2; kb3 = kn3; }
  }

  // epilogue: normalize and write bf16 [B, T, C] with heads concatenated
  const int b = bh >> 4, h = bh & 15;
#pragma unroll
  for (int f = 0; f < 4; ++f) {
#pragma unroll
    for (int i = 0; i < 8; ++i) {
      int t = t_base + mh * 8 + i;
      int d = f * 16 + nl;
      O[((size_t)(b * T_ + t)) * C_ + h * D_ + d] = (bf16)(o[f][i] / lrun[i]);
    }
  }
}

// ---------------------------------------------------------------------------
extern "C" void kernel_launch(void* const* d_in, const int* in_sizes, int n_in,
                              void* d_out, int out_size, void* d_ws,
                              size_t ws_size, hipStream_t stream) {
  (void)in_sizes; (void)n_in; (void)out_size; (void)ws_size;
  const float* x  = (const float*)d_in[0];
  const float* Wq = (const float*)d_in[1];
  const float* bq = (const float*)d_in[2];
  const float* Wk = (const float*)d_in[3];
  const float* bk = (const float*)d_in[4];
  const float* Wv = (const float*)d_in[5];
  const float* bv = (const float*)d_in[6];
  const float* Wp = (const float*)d_in[7];
  const float* bp = (const float*)d_in[8];
  float* out = (float*)d_out;

  char* w = (char*)d_ws;
  const size_t MB = (size_t)1 << 20;
  bf16* xb  = (bf16*)(w + 0);         //  8 MB: x bf16 [4096,1024]
  bf16* wqb = (bf16*)(w + 8  * MB);   //  2 MB each: weights bf16
  bf16* wkb = (bf16*)(w + 10 * MB);
  bf16* wvb = (bf16*)(w + 12 * MB);
  bf16* wpb = (bf16*)(w + 14 * MB);
  bf16* Qh  = (bf16*)(w + 16 * MB);   //  8 MB: [B,H,T,D] (pre-scaled)
  bf16* Kh  = (bf16*)(w + 24 * MB);   //  8 MB: [B,H,T,D]
  bf16* Vt  = (bf16*)(w + 32 * MB);   //  8 MB: [B,H,D,T]
  bf16* Ab  = (bf16*)(w + 40 * MB);   //  8 MB: attn bf16 [B,T,C]

  const int nx = M_ * C_;             // 4,194,304
  const int nw = C_ * C_;             // 1,048,576
  cvt_f32_bf16<<<(nx + 255) / 256, 256, 0, stream>>>(x,  xb,  nx);
  cvt_f32_bf16<<<(nw + 255) / 256, 256, 0, stream>>>(Wq, wqb, nw);
  cvt_f32_bf16<<<(nw + 255) / 256, 256, 0, stream>>>(Wk, wkb, nw);
  cvt_f32_bf16<<<(nw + 255) / 256, 256, 0, stream>>>(Wv, wvb, nw);
  cvt_f32_bf16<<<(nw + 255) / 256, 256, 0, stream>>>(Wp, wpb, nw);

  dim3 g(M_ / 128, C_ / 64);          // 32 x 16 blocks, 4 waves each
  gemm_bf16nt<<<g, 128, 0, stream>>>(xb, wqb, bq, 0.125f, 1, nullptr, Qh);
  gemm_bf16nt<<<g, 128, 0, stream>>>(xb, wkb, bk, 1.0f,   1, nullptr, Kh);
  gemm_bf16nt<<<g, 128, 0, stream>>>(xb, wvb, bv, 1.0f,   2, nullptr, Vt);

  attn_flash<<<(B_ * NH_ * (T_ / 16)) / 4, 128, 0, stream>>>(Qh, Kh, Vt, Ab);

  gemm_bf16nt<<<g, 128, 0, stream>>>(Ab, wpb, bp, 1.0f, 0, out, nullptr);
}
